// Attention_5660766896150
// MI455X (gfx1250) — matrix operand order repaired and checked by reading, not compiled, not run on hardware
//
#include <hip/hip_runtime.h>
#include <cstdint>

// ---------------- problem constants ----------------
#define BB 2
#define TT 2048
#define DD 1024
#define HH 16
#define HD 64
#define MROWS (BB * TT) // 4096

typedef __attribute__((ext_vector_type(16))) __bf16 v16bf;
typedef __attribute__((ext_vector_type(8))) float v8f;

// CDNA5 async global->LDS path (probe via __has_builtin; fallback keeps the
// round compiling if the builtin name/signature differs on this toolchain).
#if __has_builtin(__builtin_amdgcn_global_load_async_to_lds_b128) && \
    __has_builtin(__builtin_amdgcn_s_wait_asynccnt)
#define USE_ASYNC_LDS 1
#else
#define USE_ASYNC_LDS 0
#endif

#if USE_ASYNC_LDS
// Builtin prototype (from hipcc diagnostic): param0 = v4i32 __device__* (AS1),
// param1 = v4i32 __shared__* (AS3), then imm offset, imm cpol.
typedef int av4i __attribute__((ext_vector_type(4)));
#define AS1_V4I(p) ((__attribute__((address_space(1))) av4i*)(p))
#define AS3_V4I(p) ((__attribute__((address_space(3))) av4i*)(p))
#endif

// ---------------- small helpers ----------------
__device__ __forceinline__ unsigned short f2bf(float f) {
  union { float f; unsigned u; } x; x.f = f;
  unsigned r = x.u + 0x7FFFu + ((x.u >> 16) & 1u); // round-to-nearest-even
  return (unsigned short)(r >> 16);
}
__device__ __forceinline__ float bf2f(unsigned short h) {
  union { unsigned u; float f; } x; x.u = ((unsigned)h) << 16;
  return x.f;
}
__device__ __forceinline__ v8f v8f_zero() {
  v8f z;
#pragma unroll
  for (int i = 0; i < 8; ++i) z[i] = 0.0f;
  return z;
}
union FragU { uint4 u[2]; v16bf v; };
// Build a 16-half WMMA fragment from two 16B chunks (global or LDS).
__device__ __forceinline__ v16bf load_frag(const unsigned short* p0,
                                           const unsigned short* p1) {
  FragU f;
  f.u[0] = *(const uint4*)p0;
  f.u[1] = *(const uint4*)p1;
  return f.v;
}
// reductions across a 16-lane half (xor masks 1,2,4,8 stay inside the half)
__device__ __forceinline__ float half_max(float v) {
  v = fmaxf(v, __shfl_xor(v, 1, 32));
  v = fmaxf(v, __shfl_xor(v, 2, 32));
  v = fmaxf(v, __shfl_xor(v, 4, 32));
  v = fmaxf(v, __shfl_xor(v, 8, 32));
  return v;
}
__device__ __forceinline__ float half_sum(float v) {
  v += __shfl_xor(v, 1, 32);
  v += __shfl_xor(v, 2, 32);
  v += __shfl_xor(v, 4, 32);
  v += __shfl_xor(v, 8, 32);
  return v;
}

// ---------------- kernel 1a: fp32 -> bf16 ----------------
__global__ void __launch_bounds__(256)
cvt_f32_bf16(const float* __restrict__ src, unsigned short* __restrict__ dst, int n) {
  int i = blockIdx.x * blockDim.x + threadIdx.x;
  if (i < n) dst[i] = f2bf(src[i]);
}

// ---------------- kernel 1b: weights -> bf16 (wq|wk|wv concat, wo) ----------------
__global__ void __launch_bounds__(256)
cvt_weights(const float* __restrict__ wq, const float* __restrict__ wk,
            const float* __restrict__ wv, const float* __restrict__ wo,
            unsigned short* __restrict__ wcat, unsigned short* __restrict__ wob) {
  const int idx = blockIdx.x * blockDim.x + threadIdx.x; // [0, 4M)
  if (idx < 3 * 1024 * 1024) {
    const int m = idx >> 20;       // which matrix
    const int r = idx & 1048575;   // element within matrix
    const float v = (m == 0) ? wq[r] : (m == 1) ? wk[r] : wv[r];
    wcat[idx] = f2bf(v);
  } else {
    const int r = idx - 3 * 1024 * 1024;
    wob[r] = f2bf(wo[r]);
  }
}

// ---------------- WMMA GEMM: C[M,N] = A[M,K] * BT[N,K]^T ----------------
// A, BT bf16 row-major.  Each wave computes a 16x64 tile; 8 waves per WG.
// A fragment (16-bit 16x32): lane<16 row=l16 K {kb..kb+7, kb+16..kb+23}, kb=0/8.
// B fragment: lane holds column n=l16, K {kb..kb+7, kb+16..kb+23}, kb=0/8.
template <int OUTF32>
__global__ void __launch_bounds__(256)
gemm_bf16(const unsigned short* __restrict__ A, const unsigned short* __restrict__ BT,
          void* __restrict__ Cout, int N, int K, int ldc) {
  const int lane = threadIdx.x & 31;
  const int wave = threadIdx.x >> 5;
  const int half = lane >> 4;
  const int l16 = lane & 15;
  const int m0 = blockIdx.x * 16;
  const int n0 = (blockIdx.y * 8 + wave) * 64;
  if (n0 >= N) return;

  const int akb = half ? 8 : 0;
  const int bkb = half ? 8 : 0;

  v8f acc[4];
#pragma unroll
  for (int t = 0; t < 4; ++t) acc[t] = v8f_zero();

  const unsigned short* arow = A + (size_t)(m0 + l16) * K;
  const unsigned short* brow0 = BT + (size_t)(n0 + l16) * K;

  for (int k = 0; k < K; k += 32) {
    const v16bf av = load_frag(arow + k + akb, arow + k + akb + 16);
#pragma unroll
    for (int t = 0; t < 4; ++t) {
      const unsigned short* bp = brow0 + (size_t)(16 * t) * K + k + bkb;
      acc[t] = __builtin_amdgcn_wmma_f32_16x16x32_bf16(
          false, av, false, load_frag(bp, bp + 16), (short)0, acc[t], false, false);
    }
  }
#pragma unroll
  for (int t = 0; t < 4; ++t)
#pragma unroll
    for (int i = 0; i < 8; ++i) {
      const int row = m0 + i + 8 * half; // C-layout: lane<16 rows 0..7, else 8..15
      const int col = n0 + 16 * t + l16;
      if (OUTF32)
        ((float*)Cout)[(size_t)row * ldc + col] = acc[t][i];
      else
        ((unsigned short*)Cout)[(size_t)row * ldc + col] = f2bf(acc[t][i]);
    }
}

// ---------------- kernel 3: RoPE + relayout ----------------
// qkvraw [B*T, 3072] bf16 -> Q,K [B,H,T,64] bf16 (rotated), Vt [B,H,64,T] bf16
__global__ void __launch_bounds__(256)
rope_layout(const unsigned short* __restrict__ qkv, const float* __restrict__ cosT,
            const float* __restrict__ sinT, unsigned short* __restrict__ Q,
            unsigned short* __restrict__ Kd, unsigned short* __restrict__ Vt) {
  const int idx = blockIdx.x * blockDim.x + threadIdx.x; // b,h,t,d packed (2^22)
  const int d = idx & 63;
  const int t = (idx >> 6) & 2047;
  const int h = (idx >> 17) & 15;
  const int b = idx >> 21;

  const size_t rowoff = ((size_t)(b * TT + t)) * 3072 + h * HD;
  const float co = cosT[t * HD + d];
  const float si = sinT[t * HD + d];
  const int d2 = (d < 32) ? d + 32 : d - 32;
  const float sgn = (d < 32) ? -1.0f : 1.0f;

  const float qv = bf2f(qkv[rowoff + d]);
  const float qp = bf2f(qkv[rowoff + d2]);
  const float kv = bf2f(qkv[rowoff + 1024 + d]);
  const float kp = bf2f(qkv[rowoff + 1024 + d2]);
  const unsigned short vv = qkv[rowoff + 2048 + d];

  const size_t qo = ((size_t)((b * HH + h) * TT + t)) * HD + d;
  Q[qo] = f2bf(qv * co + sgn * qp * si);
  Kd[qo] = f2bf(kv * co + sgn * kp * si);
  Vt[((size_t)(b * HH + h) * HD + d) * TT + t] = vv;
}

// ---------------- kernel 4: causal flash attention ----------------
// Grid (T/128, H, B), 256 threads = 8 waves; wave owns 16 queries x 64 accum.
__global__ void __launch_bounds__(256)
flash_attn(const unsigned short* __restrict__ Q, const unsigned short* __restrict__ Kd,
           const unsigned short* __restrict__ Vt, unsigned short* __restrict__ Ob) {
  __shared__ __align__(16) unsigned short kt[32 * HD];      // 32 keys x 64
  __shared__ __align__(16) unsigned short vs[HD * 32];      // V^T: 64 x 32 keys
  __shared__ __align__(16) unsigned short ps[8 * 16 * 32];  // per-wave P tile

  const int tid = threadIdx.x;
  const int lane = tid & 31;
  const int wave = tid >> 5;
  const int half = lane >> 4;
  const int l16 = lane & 15;
  const int qb0 = blockIdx.x * 128;
  const int h = blockIdx.y;
  const int b = blockIdx.z;
  const int q0w = qb0 + wave * 16;

  const unsigned short* Qbh = Q + ((size_t)(b * HH + h) * TT) * HD;
  const unsigned short* Kbh = Kd + ((size_t)(b * HH + h) * TT) * HD;
  const unsigned short* Vbh = Vt + ((size_t)(b * HH + h) * HD) * TT;

  const int akb = half ? 8 : 0;
  const int bkb = half ? 8 : 0;

  // Q A-fragments (HD=64 -> two k=32 fragments), loaded once
  const unsigned short* qrow = Qbh + (size_t)(q0w + l16) * HD;
  const v16bf qa0 = load_frag(qrow + akb, qrow + akb + 16);
  const v16bf qa1 = load_frag(qrow + 32 + akb, qrow + 32 + akb + 16);

  float mrow[8], lrow[8];
  v8f o[4];
#pragma unroll
  for (int i = 0; i < 8; ++i) { mrow[i] = -1e30f; lrow[i] = 0.0f; }
#pragma unroll
  for (int t = 0; t < 4; ++t) o[t] = v8f_zero();

  // staging indices: 256 threads x 16B cover each 4KB tile exactly
  const int kr = tid >> 3, kc = (tid & 7) * 8; // K tile 32x64 halves
  const int vr = tid >> 2, vc = (tid & 3) * 8; // V tile 64x32 halves

  const int jend = qb0 + 128;
  for (int j = 0; j < jend; j += 32) {
#if USE_ASYNC_LDS
    __builtin_amdgcn_global_load_async_to_lds_b128(
        AS1_V4I(Kbh + (size_t)(j + kr) * HD + kc),
        AS3_V4I(&kt[kr * HD + kc]), 0, 0);
    __builtin_amdgcn_global_load_async_to_lds_b128(
        AS1_V4I(Vbh + (size_t)vr * TT + j + vc),
        AS3_V4I(&vs[vr * 32 + vc]), 0, 0);
    __builtin_amdgcn_s_wait_asynccnt(0);
#else
    *(uint4*)&kt[kr * HD + kc] = *(const uint4*)(Kbh + (size_t)(j + kr) * HD + kc);
    *(uint4*)&vs[vr * 32 + vc] = *(const uint4*)(Vbh + (size_t)vr * TT + j + vc);
#endif
    __syncthreads();

    if (j <= q0w + 15) { // wave-uniform causal skip
      // S = Q K^T for two 16-key subtiles
      v8f s[2];
#pragma unroll
      for (int sub = 0; sub < 2; ++sub) {
        v8f c = v8f_zero();
        const unsigned short* kb0 = &kt[(16 * sub + l16) * HD + bkb];
        const unsigned short* kb1 = &kt[(16 * sub + l16) * HD + 32 + bkb];
        c = __builtin_amdgcn_wmma_f32_16x16x32_bf16(
            false, qa0, false, load_frag(kb0, kb0 + 16), (short)0, c, false, false);
        c = __builtin_amdgcn_wmma_f32_16x16x32_bf16(
            false, qa1, false, load_frag(kb1, kb1 + 16), (short)0, c, false, false);
        s[sub] = c;
      }
      // online softmax over the 32-key block
#pragma unroll
      for (int i = 0; i < 8; ++i) {
        const int row = q0w + i + 8 * half;
        float s0 = (j + l16 <= row) ? s[0][i] * 0.125f : -1e30f;
        float s1 = (j + 16 + l16 <= row) ? s[1][i] * 0.125f : -1e30f;
        float rm = half_max(fmaxf(s0, s1));
        float mn = fmaxf(mrow[i], rm);
        float corr = __expf(mrow[i] - mn);
        float p0 = __expf(s0 - mn);
        float p1 = __expf(s1 - mn);
        float rs = half_sum(p0 + p1);
        lrow[i] = lrow[i] * corr + rs;
        mrow[i] = mn;
#pragma unroll
        for (int t = 0; t < 4; ++t) o[t][i] *= corr;
        const int rl = i + 8 * half;
        ps[wave * 512 + rl * 32 + l16] = f2bf(p0);
        ps[wave * 512 + rl * 32 + 16 + l16] = f2bf(p1);
      }
      // O += P V  (A-frag of P re-read from this wave's LDS region)
      const unsigned short* pw = &ps[wave * 512 + l16 * 32];
      const v16bf pa = load_frag(pw + akb, pw + akb + 16);
#pragma unroll
      for (int t = 0; t < 4; ++t) {
        const unsigned short* vrow = &vs[(16 * t + l16) * 32 + bkb];
        o[t] = __builtin_amdgcn_wmma_f32_16x16x32_bf16(
            false, pa, false, load_frag(vrow, vrow + 16), (short)0, o[t], false, false);
      }
    }
    __syncthreads();
  }

  float inv[8];
#pragma unroll
  for (int i = 0; i < 8; ++i) inv[i] = 1.0f / lrow[i];
#pragma unroll
  for (int t = 0; t < 4; ++t)
#pragma unroll
    for (int i = 0; i < 8; ++i) {
      const int row = q0w + i + 8 * half;
      const size_t off = ((size_t)(b * TT) + row) * DD + h * HD + 16 * t + l16;
      Ob[off] = f2bf(o[t][i] * inv[i]);
    }
}

// ---------------- host-side launcher ----------------
extern "C" void kernel_launch(void* const* d_in, const int* in_sizes, int n_in,
                              void* d_out, int out_size, void* d_ws, size_t ws_size,
                              hipStream_t stream) {
  (void)in_sizes; (void)n_in; (void)out_size; (void)ws_size;
  const float* x = (const float*)d_in[0];
  const float* wq = (const float*)d_in[1];
  const float* wk = (const float*)d_in[2];
  const float* wv = (const float*)d_in[3];
  const float* wo = (const float*)d_in[4];
  const float* cosT = (const float*)d_in[5];
  const float* sinT = (const float*)d_in[6];
  // d_in[7] = mask: unused, causal mask is computed analytically.

  // workspace layout (ushort elements); total 72 MB
  unsigned short* W = (unsigned short*)d_ws;
  unsigned short* xb = W;                      // x bf16          [4096,1024]
  unsigned short* wcat = W + 4194304;          // wq|wk|wv bf16   [3072,1024]
  unsigned short* wob = W + 7340032;           // wo bf16         [1024,1024]
  unsigned short* qkvraw = W + 8388608;        // x@Wcat^T bf16   [4096,3072]
  unsigned short* Qb = W + 20971520;           // [B,H,T,64]
  unsigned short* Kb = W + 25165824;           // [B,H,T,64]
  unsigned short* Vtb = W + 29360128;          // [B,H,64,T]
  unsigned short* Ob = W + 33554432;           // attn out bf16   [4096,1024]

  cvt_f32_bf16<<<16384, 256, 0, stream>>>(x, xb, 4194304);
  cvt_weights<<<16384, 256, 0, stream>>>(wq, wk, wv, wo, wcat, wob);
  gemm_bf16<0><<<dim3(MROWS / 16, 3072 / 512), 256, 0, stream>>>(
      xb, wcat, qkvraw, 3072, DD, 3072);
  rope_layout<<<16384, 256, 0, stream>>>(qkvraw, cosT, sinT, Qb, Kb, Vtb);
  flash_attn<<<dim3(TT / 128, HH, BB), 256, 0, stream>>>(Qb, Kb, Vtb, Ob);
  gemm_bf16<1><<<dim3(MROWS / 16, DD / 512), 256, 0, stream>>>(
      Ob, wob, (float*)d_out, DD, DD, DD);
}
